// Head_24953759989982
// MI455X (gfx1250) — matrix operand order repaired and checked by reading, not compiled, not run on hardware
//
#include <hip/hip_runtime.h>

// ---------------------------------------------------------------------------
// Fused causal single-head attention for MI455X (gfx1250, wave32, WMMA).
// B=1024, T=128, C=1024, H=64.  One workgroup (8 waves) per batch element.
//   * Projections + attention on v_wmma_f32_16x16x32_bf16, f32 accumulate.
//   * x tiles DMA'd into LDS with global_load_async_to_lds_b128 (ASYNCcnt),
//     double-buffered so HBM latency overlaps WMMA.
//   * Weights pre-packed once to bf16, transposed K-contiguous, served from
//     WGP$/L2 directly as B fragments (global_load_b128, s_clause'd).
//   * All LDS fragment reads are 16-byte ds_load_b128.
//   * Causal mask folded into S-tile store; softmax is 2 vectorized passes;
//     P stored unnormalized, 1/sum folded into O after the PV matmul.
// HBM traffic: x 512MB + out 32MB  ->  ~23 us floor at 23.3 TB/s.
// ---------------------------------------------------------------------------

typedef __attribute__((ext_vector_type(16))) __bf16        v16bf;
typedef __attribute__((ext_vector_type(8)))  float         v8f;
typedef __attribute__((ext_vector_type(8)))  unsigned int  v8u;

#define BB 1024
#define TT 128
#define CC 1024
#define HH 64

__device__ __forceinline__ unsigned pack_bf16(float a, float b) {
  unsigned ua = __builtin_bit_cast(unsigned, a) + 0x8000u;
  unsigned ub = __builtin_bit_cast(unsigned, b) + 0x8000u;
  return __builtin_amdgcn_perm(ub, ua, 0x07060302u);  // lo16=hi(a), hi16=hi(b)
}
__device__ __forceinline__ unsigned short f2bf(float f) {
  unsigned u = __builtin_bit_cast(unsigned, f) + 0x8000u;
  return (unsigned short)(u >> 16);
}
__device__ __forceinline__ v16bf frag_cast(v8u r) {
  return __builtin_bit_cast(v16bf, r);
}

// CDNA5 async DMA: global -> LDS, 16B per lane, tracked by ASYNCcnt.
__device__ __forceinline__ void async_load_b128(unsigned lds_off, const void* g) {
  asm volatile("global_load_async_to_lds_b128 %0, %1, off"
               :: "v"(lds_off), "v"((unsigned long long)g) : "memory");
}
__device__ __forceinline__ void wait_async0() {
  asm volatile("s_wait_asynccnt 0x0" ::: "memory");
}
// Zero-cost compiler fence: wave-private LDS ordering is guaranteed by the
// in-order DScnt pipeline; we only need to stop compiler reordering.
__device__ __forceinline__ void wave_fence() { asm volatile("" ::: "memory"); }

// ---------------------------------------------------------------------------
// Prologue: pack Wq/Wk/Wv [1024][64] f32 -> wsT [3][64][1024] bf16 (K-contig).
// ---------------------------------------------------------------------------
__global__ __launch_bounds__(256)
void Head_w_pack(const float* __restrict__ Wk, const float* __restrict__ Wq,
                 const float* __restrict__ Wv, unsigned short* __restrict__ wsT) {
  int m = blockIdx.y;                                  // 0=Q, 1=K, 2=V
  const float* src = (m == 0) ? Wq : ((m == 1) ? Wk : Wv);
  unsigned short* dst = wsT + (size_t)m * HH * CC;
  int idx = blockIdx.x * 256 + threadIdx.x;            // 0..65535
  int k = idx >> 6;
  int n = idx & 63;
  dst[(size_t)n * CC + k] = f2bf(src[(size_t)k * HH + n]);
}

// ---------------------------------------------------------------------------
// Main fused kernel.
// ---------------------------------------------------------------------------
__global__ __launch_bounds__(256)
void Head_attn_fused(const float* __restrict__ x,
                     const unsigned short* __restrict__ wsT,
                     float* __restrict__ out) {
  extern __shared__ char smem[];
  // Persistent region (48 KB):
  unsigned short* qlds  = (unsigned short*)(smem);             // [128][64] bf16
  unsigned short* klds  = (unsigned short*)(smem + 16 * 1024); // [128][64] bf16
  unsigned short* vTlds = (unsigned short*)(smem + 32 * 1024); // [64][128] bf16
  // Overlapped region (96 KB) + 512B:
  char* regA = smem + 48 * 1024;
  float* xbuf[2] = { (float*)regA, (float*)(regA + 16 * 1024) }; // [128][32] f32 x2
  float*          slds   = (float*)regA;                         // [8][16][128] f32
  unsigned short* plds   = (unsigned short*)(regA + 64 * 1024);  // [8][16][128] bf16
  float*          rscale = (float*)(regA + 96 * 1024);           // [128] 1/rowsum

  const int b    = blockIdx.x;
  const int tid  = threadIdx.x;
  const int w    = tid >> 5;      // wave 0..7 (wave32)
  const int lane = tid & 31;
  const int ln   = lane & 15;
  const int hi   = lane >> 4;

  const float* xb = x + (size_t)b * TT * CC;
  const unsigned xls[2] = { (unsigned)(unsigned long long)(void*)xbuf[0],
                            (unsigned)(unsigned long long)(void*)xbuf[1] };

  // Per-thread DMA slice of a 128x32 f32 tile: 4 x b128.
  const int srow = tid >> 3;
  const int scol = (tid & 7) << 2;

  const v8f vzero = {0.f, 0.f, 0.f, 0.f, 0.f, 0.f, 0.f, 0.f};
  v8f accQ[4], accK[4], accV[4];
#pragma unroll
  for (int i = 0; i < 4; ++i) { accQ[i] = vzero; accK[i] = vzero; accV[i] = vzero; }

  // Issue DMA for K-step 0.
#pragma unroll
  for (int i = 0; i < 4; ++i) {
    int row = srow + i * 32;
    async_load_b128(xls[0] + (unsigned)((row * 32 + scol) * 4),
                    xb + (size_t)row * CC + scol);
  }

  const unsigned int* wsT32 = (const unsigned int*)wsT;  // [3*64][512] K-pairs

  // ======================= Phase 1: Q/K/V projections ======================
  for (int kk = 0; kk < CC / 32; ++kk) {
    const int buf = kk & 1;
    wait_async0();        // this wave's DMA for tile kk complete
    __syncthreads();      // all waves' DMA complete; all reads of kk-1 done
    if (kk + 1 < CC / 32) {
#pragma unroll
      for (int i = 0; i < 4; ++i) {   // prefetch tile kk+1 into other buffer
        int row = srow + i * 32;
        async_load_b128(xls[buf ^ 1] + (unsigned)((row * 32 + scol) * 4),
                        xb + (size_t)row * CC + (kk + 1) * 32 + scol);
      }
    }

    // ---- A fragment: 4x ds_load_b128 (f32) + v_perm bf16 packing ----
    const float* xrow = xbuf[buf] + (16 * w + ln) * 32;
    float4 f0 = *(const float4*)(xrow + hi * 8);
    float4 f1 = *(const float4*)(xrow + hi * 8 + 4);
    float4 f2 = *(const float4*)(xrow + 16 + hi * 8);
    float4 f3 = *(const float4*)(xrow + 16 + hi * 8 + 4);
    v8u araw;
    araw[0] = pack_bf16(f0.x, f0.y); araw[1] = pack_bf16(f0.z, f0.w);
    araw[2] = pack_bf16(f1.x, f1.y); araw[3] = pack_bf16(f1.z, f1.w);
    araw[4] = pack_bf16(f2.x, f2.y); araw[5] = pack_bf16(f2.z, f2.w);
    araw[6] = pack_bf16(f3.x, f3.y); araw[7] = pack_bf16(f3.z, f3.w);
    v16bf afrag = frag_cast(araw);

    // ---- 12 WMMAs; B fragments straight from WGP$-resident packed W ----
#pragma unroll
    for (int m = 0; m < 3; ++m) {
#pragma unroll
      for (int nt = 0; nt < 4; ++nt) {
        const uint4* bp = (const uint4*)(wsT32 +
            ((size_t)(m * 64 + nt * 16 + ln) * 512 + kk * 16 + hi * 4));
        uint4 b0 = bp[0], b1 = bp[2];
        v8u braw;
        braw[0] = b0.x; braw[1] = b0.y; braw[2] = b0.z; braw[3] = b0.w;
        braw[4] = b1.x; braw[5] = b1.y; braw[6] = b1.z; braw[7] = b1.w;
        v16bf bfrag = frag_cast(braw);
        if (m == 0)
          accQ[nt] = __builtin_amdgcn_wmma_f32_16x16x32_bf16(false, afrag, false, bfrag,
                                                             (short)0, accQ[nt], false, false);
        else if (m == 1)
          accK[nt] = __builtin_amdgcn_wmma_f32_16x16x32_bf16(false, afrag, false, bfrag,
                                                             (short)0, accK[nt], false, false);
        else
          accV[nt] = __builtin_amdgcn_wmma_f32_16x16x32_bf16(false, afrag, false, bfrag,
                                                             (short)0, accV[nt], false, false);
      }
    }
  }
  __syncthreads();  // x buffers dead; region reused as S/P below

  // ---- write q, k row-major; v transposed (bf16) to LDS ----
#pragma unroll
  for (int nt = 0; nt < 4; ++nt) {
#pragma unroll
    for (int r = 0; r < 8; ++r) {
      int m   = r + hi * 8;
      int row = 16 * w + m;
      int n   = nt * 16 + ln;
      qlds[row * HH + n]  = f2bf(accQ[nt][r]);
      klds[row * HH + n]  = f2bf(accK[nt][r]);
      vTlds[n * TT + row] = f2bf(accV[nt][r]);
    }
  }
  __syncthreads();

  // ============ Phase 2: S = Q K^T, causal mask folded into store ==========
  const unsigned int* q32 = (const unsigned int*)qlds;   // [128][32] pairs
  const unsigned int* k32 = (const unsigned int*)klds;   // [128][32] pairs
  float* srow_w = slds + w * 16 * TT;                    // wave-private [16][128]
  const float NEG_INF = -__builtin_inff();
#pragma unroll
  for (int st = 0; st < 8; ++st) {
    v8f acc = vzero;
#pragma unroll
    for (int hs = 0; hs < 2; ++hs) {
      const uint4* qp = (const uint4*)(q32 + (16 * w + ln) * 32 + hi * 4 + hs * 16);
      uint4 qa = qp[0], qb = qp[2];
      const uint4* kp = (const uint4*)(k32 + (st * 16 + ln) * 32 + hi * 4 + hs * 16);
      uint4 kb0 = kp[0], kb1 = kp[2];
      v8u araw, braw;
      araw[0] = qa.x; araw[1] = qa.y; araw[2] = qa.z; araw[3] = qa.w;
      araw[4] = qb.x; araw[5] = qb.y; araw[6] = qb.z; araw[7] = qb.w;
      braw[0] = kb0.x; braw[1] = kb0.y; braw[2] = kb0.z; braw[3] = kb0.w;
      braw[4] = kb1.x; braw[5] = kb1.y; braw[6] = kb1.z; braw[7] = kb1.w;
      acc = __builtin_amdgcn_wmma_f32_16x16x32_bf16(false, frag_cast(araw), false,
                                                    frag_cast(braw), (short)0, acc,
                                                    false, false);
    }
#pragma unroll
    for (int r = 0; r < 8; ++r) {
      int m = r + hi * 8;
      int t = 16 * w + m;
      int s = st * 16 + ln;
      srow_w[m * TT + s] = (s <= t) ? acc[r] : NEG_INF;   // mask here, once
    }
  }
  wave_fence();  // wave-private S; LDS in-order per wave, no block barrier

  // ============ Softmax: 2 vectorized passes, unnormalized P ===============
  {
    int m = ln;                                  // query row within wave tile
    const float4* row4 = (const float4*)(srow_w + m * TT);
    int q0 = hi * 16;                            // each lane-half: 16 float4s
    float mx = NEG_INF;
    for (int i = 0; i < 16; ++i) {
      float4 v = row4[q0 + i];
      mx = fmaxf(mx, fmaxf(fmaxf(v.x, v.y), fmaxf(v.z, v.w)));
    }
    mx = fmaxf(mx, __shfl_xor(mx, 16, 32));      // finite: s<=t region non-empty
    float sum = 0.f;
    unsigned int* prow32 = (unsigned int*)(plds + (w * 16 + m) * TT);
    for (int i = 0; i < 16; ++i) {
      float4 v = row4[q0 + i];
      float e0 = __expf(v.x - mx), e1 = __expf(v.y - mx);
      float e2 = __expf(v.z - mx), e3 = __expf(v.w - mx);
      sum += (e0 + e1) + (e2 + e3);
      prow32[(q0 + i) * 2 + 0] = pack_bf16(e0, e1);
      prow32[(q0 + i) * 2 + 1] = pack_bf16(e2, e3);
    }
    sum += __shfl_xor(sum, 16, 32);
    if (hi == 0) rscale[w * 16 + m] = 1.f / sum; // normalize after PV instead
  }
  wave_fence();

  // ============ O = (P @ V) * (1/rowsum) ===================================
  const unsigned int* p32 = (const unsigned int*)(plds + w * 16 * TT); // [16][64] pairs
  const unsigned int* v32 = (const unsigned int*)vTlds;                // [64][64] pairs
  const float4 rs0 = *(const float4*)(rscale + w * 16 + hi * 8);
  const float4 rs1 = *(const float4*)(rscale + w * 16 + hi * 8 + 4);
  const float rs[8] = { rs0.x, rs0.y, rs0.z, rs0.w, rs1.x, rs1.y, rs1.z, rs1.w };
  float* ob = out + (size_t)b * TT * HH;
#pragma unroll
  for (int nt = 0; nt < 4; ++nt) {
    v8f acc = vzero;
#pragma unroll
    for (int ks = 0; ks < 4; ++ks) {
      const uint4* pp = (const uint4*)(p32 + ln * 64 + hi * 4 + ks * 16);
      uint4 pa = pp[0], pb = pp[2];
      const uint4* vp = (const uint4*)(v32 + (nt * 16 + ln) * 64 + hi * 4 + ks * 16);
      uint4 vb0 = vp[0], vb1 = vp[2];
      v8u araw, braw;
      araw[0] = pa.x; araw[1] = pa.y; araw[2] = pa.z; araw[3] = pa.w;
      araw[4] = pb.x; araw[5] = pb.y; araw[6] = pb.z; araw[7] = pb.w;
      braw[0] = vb0.x; braw[1] = vb0.y; braw[2] = vb0.z; braw[3] = vb0.w;
      braw[4] = vb1.x; braw[5] = vb1.y; braw[6] = vb1.z; braw[7] = vb1.w;
      acc = __builtin_amdgcn_wmma_f32_16x16x32_bf16(false, frag_cast(araw), false,
                                                    frag_cast(braw), (short)0, acc,
                                                    false, false);
    }
#pragma unroll
    for (int r = 0; r < 8; ++r) {
      int m = r + hi * 8;
      ob[(size_t)(16 * w + m) * HH + nt * 16 + ln] = acc[r] * rs[r];
    }
  }
}

extern "C" void kernel_launch(void* const* d_in, const int* in_sizes, int n_in,
                              void* d_out, int out_size, void* d_ws, size_t ws_size,
                              hipStream_t stream) {
  (void)in_sizes; (void)n_in; (void)out_size; (void)ws_size;
  const float* x  = (const float*)d_in[0];
  const float* Wk = (const float*)d_in[1];
  const float* Wq = (const float*)d_in[2];
  const float* Wv = (const float*)d_in[3];
  float* out = (float*)d_out;
  unsigned short* wsT = (unsigned short*)d_ws;   // 3*64*1024*2 = 384 KB

  Head_w_pack<<<dim3(256, 3), dim3(256), 0, stream>>>(Wk, Wq, Wv, wsT);

  const size_t shmem = 148 * 1024;  // 48KB qkv + 96KB (xbuf/S/P) + rscale
  Head_attn_fused<<<dim3(BB), dim3(256), shmem, stream>>>(x, wsT, out);
}
